// DIEN_79448305041886
// MI455X (gfx1250) — matrix operand order repaired and checked by reading, not compiled, not run on hardware
//
#include <hip/hip_runtime.h>

// ---------------------------------------------------------------------------
// DIEN (GRU -> v_proj "attention" -> AUGRU) for MI455X / gfx1250, wave32.
//
//   * non-recurrent matmuls hoisted into big bf16-WMMA GEMMs (fp32 accum)
//   * double-buffered LDS B-tile, register-staged global->LDS copy,
//     one barrier per K-step, batched ds_load -> 4..8 back-to-back WMMAs
//   * bulk GEMMs: 256M x 64N per block (MR=2); recurrence GEMMs: 128M (MR=1)
//   * fp32 pre-activation streams (Xih/AugGi, ~315 MB each > 192 MB L2) use
//     non-temporal hints; bf16 weights/activations stay cache-resident
//
// Workspace requirement: ~800 MB.
// ---------------------------------------------------------------------------

namespace {

constexpr int Bn = 256;   // batch
constexpr int Tn = 200;   // seq len
constexpr int Hn = 512;   // hidden

typedef __attribute__((ext_vector_type(16))) __bf16 v16bf;
typedef __attribute__((ext_vector_type(8)))  __bf16 v8bf;
typedef __attribute__((ext_vector_type(8)))  float  v8f;

union BF16x16 { v16bf v; v8bf h[2]; };

__device__ inline v16bf make_frag(v8bf lo, v8bf hi) {
  BF16x16 u; u.h[0] = lo; u.h[1] = hi; return u.v;
}

// A fragment: 16x32 bf16, row-major source. Lane L covers row M = L%16.
// Lanes 0-15: elems 0..7 = K0..7, elems 8..15 = K16..23.
// Lanes 16-31: elems 0..7 = K8..15, elems 8..15 = K24..31.
__device__ inline v16bf load_a_frag(const __bf16* a_tile, int lda, int lane) {
  const int m    = lane & 15;
  const int half = lane >> 4;
  const __bf16* p = a_tile + m * lda + half * 8;
  return make_frag(*(const v8bf*)(p), *(const v8bf*)(p + 16));
}

// B fragment from an LDS tile stored as [64 rows n][32 k] bf16 (row = output
// column n of W, W being [N][K] row-major). Lane L covers column N = L%16;
// lanes 0-15 hold K0..7,K16..23; lanes 16-31 hold K8..15,K24..31.
__device__ inline v16bf load_b_frag_lds(const __bf16* bs, int lane) {
  const int n    = lane & 15;
  const int half = lane >> 4;
  const __bf16* p = bs + n * 32 + half * 8;
  return make_frag(*(const v8bf*)(p), *(const v8bf*)(p + 16));
}

// C[M,N] (+= / =) A[M,K](bf16) @ W[N,K]^T(bf16) + bias[N]
// grid = (N/64, M/(128*MR)); block = 256 threads = 8 waves.
// Each wave: MR*16 M-rows x 64 N-cols. B tile double-buffered through LDS.
template <int MR, bool STORE_BF16, bool ACCUM>
__global__ __launch_bounds__(256) void gemm_wmma_bf16(
    const __bf16* __restrict__ A, int lda,
    const __bf16* __restrict__ W, int ldw,
    const float*  __restrict__ bias,
    float* __restrict__ Cf, __bf16* __restrict__ Cb,
    int ldc, int K) {
  __shared__ __attribute__((aligned(16))) __bf16 Bs[2][64 * 32];

  const int lane = threadIdx.x & 31;
  const int wave = threadIdx.x >> 5;
  const int n0   = blockIdx.x * 64;
  const int m0   = blockIdx.y * (128 * MR) + wave * (16 * MR);

  v8f acc[MR][4];
#pragma unroll
  for (int s = 0; s < 4; ++s) {
    const float bv = bias ? bias[n0 + s * 16 + (lane & 15)] : 0.0f;
#pragma unroll
    for (int r = 0; r < MR; ++r)
#pragma unroll
      for (int i = 0; i < 8; ++i) acc[r][s][i] = bv;
  }

  // Cooperative B-tile staging: 64x32 bf16 = 4 KB; 16 B per thread.
  const int nrow = threadIdx.x >> 2;        // 0..63
  const int kk   = (threadIdx.x & 3) * 8;   // 0,8,16,24
  const __bf16* wp = W + (size_t)(n0 + nrow) * ldw + kk;

  const int nsteps = K >> 5;
  // Prologue: tile 0 -> LDS buffer 0; tile 1 staged in registers.
  *(v8bf*)(&Bs[0][nrow * 32 + kk]) = *(const v8bf*)(wp);
  v8bf stage{};
  if (nsteps > 1) stage = *(const v8bf*)(wp + 32);
  __syncthreads();

  for (int i = 0; i < nsteps; ++i) {
    const int k0 = i << 5;
    const __bf16* bs = &Bs[i & 1][0];

    // Batch all LDS B-fragment reads first (one dscnt wait, then WMMAs).
    v16bf bfr[4];
#pragma unroll
    for (int s = 0; s < 4; ++s) bfr[s] = load_b_frag_lds(bs + s * 16 * 32, lane);

#pragma unroll
    for (int r = 0; r < MR; ++r) {
      const v16bf a =
          load_a_frag(A + (size_t)(m0 + r * 16) * lda + k0, lda, lane);
#pragma unroll
      for (int s = 0; s < 4; ++s)
        acc[r][s] = __builtin_amdgcn_wmma_f32_16x16x32_bf16(
            false, a, false, bfr[s], (short)0, acc[r][s], false, false);
    }

    if (i + 1 < nsteps) {
      // Buffer (i+1)&1 was last read at step i-1; barrier at end of that
      // step guarantees it is free. Store staged tile, refill stage.
      *(v8bf*)(&Bs[(i + 1) & 1][nrow * 32 + kk]) = stage;
      if (i + 2 < nsteps) {
        stage = *(const v8bf*)(wp + (i + 2) * 32);
        __builtin_prefetch(wp + (i + 3) * 32, 0, 1);
      }
      __syncthreads();
    }
  }

  // C/D layout: lane%16 = N, lane/16 selects M rows 0-7 vs 8-15, VGPR i = M+i.
  const int nn = lane & 15;
  const int mb = (lane >> 4) * 8;
#pragma unroll
  for (int r = 0; r < MR; ++r)
#pragma unroll
    for (int s = 0; s < 4; ++s) {
      const int col = n0 + s * 16 + nn;
#pragma unroll
      for (int i = 0; i < 8; ++i) {
        const size_t off = (size_t)(m0 + r * 16 + mb + i) * ldc + col;
        float v = acc[r][s][i];
        if (ACCUM) v += __builtin_nontemporal_load(&Cf[off]);
        if (STORE_BF16) Cb[off] = (__bf16)v;
        else            __builtin_nontemporal_store(v, &Cf[off]);
      }
    }
}

__device__ inline float sigmoidf_(float x) { return 1.0f / (1.0f + __expf(-x)); }
__device__ inline float tanhf_(float x) {
  const float e = __expf(-2.0f * x);
  return (1.0f - e) / (1.0f + e);
}

// PyTorch GRUCell gate math. gi already contains bih, gh already contains bhh.
__global__ __launch_bounds__(256) void gru_elem(
    const float* __restrict__ gi,   // [B,3H] slice (input projection)
    const float* __restrict__ gh,   // [B,3H]       (hidden projection)
    const float* h_in,              // [B,H] previous hidden (fp32)
    float* h_out,                   // [B,H] new hidden (fp32)
    __bf16* __restrict__ h_bf_out)  // [B,H] new hidden (bf16, feeds GEMMs)
{
  const int idx = blockIdx.x * blockDim.x + threadIdx.x;
  if (idx >= Bn * Hn) return;
  const int b = idx / Hn;
  const int j = idx - b * Hn;
  const float* gib = gi + (size_t)b * 3 * Hn;
  const float* ghb = gh + (size_t)b * 3 * Hn;
  const float r = sigmoidf_(gib[j] + ghb[j]);
  const float z = sigmoidf_(gib[j + Hn] + ghb[j + Hn]);
  const float n = tanhf_(gib[j + 2 * Hn] + r * ghb[j + 2 * Hn]);
  const float h = h_in[idx];
  const float hn = (1.0f - z) * n + z * h;
  h_out[idx] = hn;
  h_bf_out[idx] = (__bf16)hn;
}

__global__ void cvt_f32_bf16(const float* __restrict__ in,
                             __bf16* __restrict__ out, int n) {
  for (int i = blockIdx.x * blockDim.x + threadIdx.x; i < n;
       i += gridDim.x * blockDim.x)
    out[i] = (__bf16)in[i];
}

__global__ void zero_f32(float* p, int n) {
  for (int i = blockIdx.x * blockDim.x + threadIdx.x; i < n;
       i += gridDim.x * blockDim.x)
    p[i] = 0.0f;
}

__global__ void zero_bf16(__bf16* p, int n) {
  for (int i = blockIdx.x * blockDim.x + threadIdx.x; i < n;
       i += gridDim.x * blockDim.x)
    p[i] = (__bf16)0.0f;
}

}  // namespace

extern "C" void kernel_launch(void* const* d_in, const int* in_sizes, int n_in,
                              void* d_out, int out_size, void* d_ws,
                              size_t ws_size, hipStream_t stream) {
  (void)in_sizes; (void)n_in; (void)out_size; (void)ws_size;

  const float* seq      = (const float*)d_in[0];
  // d_in[1] target_emb, d_in[6..9] q/k proj: mathematically dead (singleton softmax)
  const float* gru_Wih  = (const float*)d_in[2];
  const float* gru_Whh  = (const float*)d_in[3];
  const float* gru_bih  = (const float*)d_in[4];
  const float* gru_bhh  = (const float*)d_in[5];
  const float* v_W      = (const float*)d_in[10];
  const float* v_b      = (const float*)d_in[11];
  const float* aug_Wih  = (const float*)d_in[12];
  const float* aug_Whh  = (const float*)d_in[13];
  const float* aug_bih  = (const float*)d_in[14];
  const float* aug_bhh  = (const float*)d_in[15];

  // ---- workspace bump allocation (256B aligned) ----
  char* ws = (char*)d_ws;
  auto balloc = [&](size_t bytes) {
    char* p = ws;
    ws += (bytes + 255) & ~(size_t)255;
    return p;
  };
  const size_t nW_gih = 3 * Hn * Hn;       // 786432
  const size_t nW_ghh = 3 * Hn * Hn;
  const size_t nW_v   = (size_t)Hn * Hn;   // 262144
  const size_t nW_aih = 3 * Hn * 2 * Hn;   // 1572864
  const size_t nW_ahh = 3 * Hn * Hn;
  const size_t nX     = (size_t)Bn * Tn * Hn;       // 26,214,400
  const size_t nXih   = (size_t)Bn * Tn * 3 * Hn;   // 78,643,200
  const size_t nGh    = (size_t)Bn * 3 * Hn;
  const size_t nHF    = (size_t)Bn * Hn;

  __bf16* gWih_b = (__bf16*)balloc(nW_gih * 2);
  __bf16* gWhh_b = (__bf16*)balloc(nW_ghh * 2);
  __bf16* vW_b   = (__bf16*)balloc(nW_v   * 2);
  __bf16* aWih_b = (__bf16*)balloc(nW_aih * 2);
  __bf16* aWhh_b = (__bf16*)balloc(nW_ahh * 2);
  __bf16* Xbf    = (__bf16*)balloc(nX * 2);
  float*  Xih    = (float*) balloc(nXih * 4);
  __bf16* Gbf    = (__bf16*)balloc(nX * 2);   // all g_t, bf16
  __bf16* Attnb  = (__bf16*)balloc(nX * 2);
  float*  AugGi  = (float*) balloc(nXih * 4);
  float*  Gh     = (float*) balloc(nGh * 4);  // per-step hidden projection
  float*  Hgru_f = (float*) balloc(nHF * 4);
  float*  Haug_f = (float*) balloc(nHF * 4);
  __bf16* Haug_b = (__bf16*)balloc(nHF * 2);
  __bf16* Hzero  = (__bf16*)balloc(nHF * 2);  // bf16 zeros (t==0 hidden)

  const dim3 blk(256);
  auto gs = [](size_t n) { return dim3((unsigned)((n + 255) / 256)); };

  // ---- stage 0: fp32 -> bf16 conversions + state init ----
  cvt_f32_bf16<<<gs(nW_gih), blk, 0, stream>>>(gru_Wih, gWih_b, (int)nW_gih);
  cvt_f32_bf16<<<gs(nW_ghh), blk, 0, stream>>>(gru_Whh, gWhh_b, (int)nW_ghh);
  cvt_f32_bf16<<<gs(nW_v),   blk, 0, stream>>>(v_W,     vW_b,   (int)nW_v);
  cvt_f32_bf16<<<gs(nW_aih), blk, 0, stream>>>(aug_Wih, aWih_b, (int)nW_aih);
  cvt_f32_bf16<<<gs(nW_ahh), blk, 0, stream>>>(aug_Whh, aWhh_b, (int)nW_ahh);
  cvt_f32_bf16<<<gs(nX),     blk, 0, stream>>>(seq,     Xbf,    (int)nX);
  zero_f32 <<<gs(nHF), blk, 0, stream>>>(Hgru_f, (int)nHF);
  zero_f32 <<<gs(nHF), blk, 0, stream>>>(Haug_f, (int)nHF);
  zero_bf16<<<gs(nHF), blk, 0, stream>>>(Haug_b, (int)nHF);
  zero_bf16<<<gs(nHF), blk, 0, stream>>>(Hzero,  (int)nHF);

  // ---- stage 1: Xih = seq @ gru_Wih^T + bih   [51200 x 1536, K=512] ----
  gemm_wmma_bf16<2, false, false>
      <<<dim3(3 * Hn / 64, Bn * Tn / 256), blk, 0, stream>>>(
          Xbf, Hn, gWih_b, Hn, gru_bih, Xih, nullptr, 3 * Hn, Hn);

  // ---- stage 2: GRU recurrence (sequential, 200 steps) ----
  for (int t = 0; t < Tn; ++t) {
    const __bf16* hprev = (t == 0) ? Hzero : (Gbf + (size_t)(t - 1) * Bn * Hn);
    gemm_wmma_bf16<1, false, false>
        <<<dim3(3 * Hn / 64, Bn / 128), blk, 0, stream>>>(
            hprev, Hn, gWhh_b, Hn, gru_bhh, Gh, nullptr, 3 * Hn, Hn);
    gru_elem<<<gs(nHF), blk, 0, stream>>>(
        Xih + (size_t)t * Bn * 3 * Hn, Gh, Hgru_f, Hgru_f,
        Gbf + (size_t)t * Bn * Hn);
  }

  // ---- stage 3: Attn = G @ v_W^T + v_b   [51200 x 512, K=512] -> bf16 ----
  gemm_wmma_bf16<2, true, false>
      <<<dim3(Hn / 64, Bn * Tn / 256), blk, 0, stream>>>(
          Gbf, Hn, vW_b, Hn, v_b, nullptr, Attnb, Hn, Hn);

  // ---- stage 4: AugGi = [G|Attn] @ aug_Wih^T + bih, as two K=512 GEMMs ----
  gemm_wmma_bf16<2, false, false>
      <<<dim3(3 * Hn / 64, Bn * Tn / 256), blk, 0, stream>>>(
          Gbf, Hn, aWih_b, 2 * Hn, aug_bih, AugGi, nullptr, 3 * Hn, Hn);
  gemm_wmma_bf16<2, false, true>
      <<<dim3(3 * Hn / 64, Bn * Tn / 256), blk, 0, stream>>>(
          Attnb, Hn, aWih_b + Hn, 2 * Hn, nullptr, AugGi, nullptr, 3 * Hn, Hn);

  // ---- stage 5: AUGRU recurrence; final step writes d_out (fp32 [B,H]) ----
  for (int t = 0; t < Tn; ++t) {
    gemm_wmma_bf16<1, false, false>
        <<<dim3(3 * Hn / 64, Bn / 128), blk, 0, stream>>>(
            Haug_b, Hn, aWhh_b, Hn, aug_bhh, Gh, nullptr, 3 * Hn, Hn);
    float* hout = (t == Tn - 1) ? (float*)d_out : Haug_f;
    gru_elem<<<gs(nHF), blk, 0, stream>>>(
        AugGi + (size_t)t * Bn * 3 * Hn, Gh, Haug_f, hout, Haug_b);
  }
}
